// GTBlock_23467701305426
// MI455X (gfx1250) — compile-verified, matrix operand order checked by reading, not run on hardware
//
#include <hip/hip_runtime.h>
#include <hip/hip_bf16.h>
#include <math.h>

// ---------------------------------------------------------------------------
// CDNA5 (gfx1250) WMMA types
// ---------------------------------------------------------------------------
typedef __attribute__((ext_vector_type(16))) __bf16 v16bf;
typedef __attribute__((ext_vector_type(8)))  __bf16 v8bf;
typedef __attribute__((ext_vector_type(4)))  __bf16 v4bf;
typedef __attribute__((ext_vector_type(8)))  float  v8f;

#define WMMA_BF16(a, b, c) \
  __builtin_amdgcn_wmma_f32_16x16x32_bf16(false, (a), false, (b), (short)0, (c), false, false)

// Convert float4 -> 4 packed bf16 (compiler emits v_cvt_pk_bf16_f32 pairs).
__device__ __forceinline__ v4bf cvt4(float4 f) {
  v4bf r;
  r[0] = (__bf16)f.x; r[1] = (__bf16)f.y; r[2] = (__bf16)f.z; r[3] = (__bf16)f.w;
  return r;
}

// Assemble a 16x32 bf16 WMMA fragment for this lane from two contiguous
// 16-byte LDS chunks (k in [8*half,8*half+8) and [16+8*half,+8), ISA 7.12.2).
// Both pointers must be 16-byte aligned -> two ds_load_b128.
__device__ __forceinline__ v16bf frag_ld(const __bf16* p0, const __bf16* p1) {
  v8bf lo = *(const v8bf*)p0;
  v8bf hi = *(const v8bf*)p1;
  return __builtin_shufflevector(lo, hi, 0, 1, 2, 3, 4, 5, 6, 7,
                                 8, 9, 10, 11, 12, 13, 14, 15);
}

// ---------------------------------------------------------------------------
// Generic bf16-in / fp32-accum WMMA GEMM: C[M,N] = A[M,K] @ B[K,N] (+bias)
// Block = 128 threads = 4 waves; block tile 64x64; wave tile 32x32; K step 32.
// M,N multiples of 64; K multiple of 32 (true for all GEMMs in this model).
// A tile row-major (k innermost); B tile TRANSPOSED in LDS (k innermost) so
// every fragment load is two ds_load_b128. Row stride 40 elems = 80 B:
// 16-B aligned and bank-conflict-free across the 16 lanes of a fragment row.
// ---------------------------------------------------------------------------
__global__ __launch_bounds__(128) void gemm_bf16_wmma(
    const float* __restrict__ A, const float* __restrict__ B,
    const float* __restrict__ bias, float* __restrict__ C,
    int M, int N, int K) {
  __shared__ __align__(16) __bf16 sA[64][40];   // [m][k]
  __shared__ __align__(16) __bf16 sBt[64][40];  // [n][k]  (B transposed)

  const int tid  = threadIdx.x;
  const int lane = tid & 31;
  const int wave = tid >> 5;
  const int wm   = (wave >> 1) * 32;
  const int wn   = (wave & 1) * 32;
  const int bm   = blockIdx.x * 64;
  const int bn   = blockIdx.y * 64;
  const int half = lane >> 4;
  const int mr   = lane & 15;

  v8f acc[2][2];
#pragma unroll
  for (int i = 0; i < 2; ++i)
#pragma unroll
    for (int j = 0; j < 2; ++j)
#pragma unroll
      for (int r = 0; r < 8; ++r) acc[i][j][r] = 0.f;

  for (int k0 = 0; k0 < K; k0 += 32) {
    if (k0 + 32 < K) {  // prefetch next tiles (global_prefetch_b8)
      __builtin_prefetch(&A[(size_t)(bm + (tid & 63)) * K + k0 + 32], 0, 1);
      __builtin_prefetch(&B[(size_t)(k0 + 32 + (tid & 31)) * N + bn], 0, 1);
    }
    // Stage A: 64x32, 4 floats per thread-iter (float4 load, b64 LDS store).
#pragma unroll
    for (int i = tid; i < 512; i += 128) {
      int r = i >> 3, qc = (i & 7) * 4;
      float4 f = *(const float4*)&A[(size_t)(bm + r) * K + k0 + qc];
      *(v4bf*)&sA[r][qc] = cvt4(f);
    }
    // Stage B transposed: column c, k-quad kq; per-t loads coalesced over c.
#pragma unroll
    for (int i = tid; i < 512; i += 128) {
      int c = i & 63, kq = (i >> 6) * 4;
      float4 f;
      f.x = B[(size_t)(k0 + kq + 0) * N + bn + c];
      f.y = B[(size_t)(k0 + kq + 1) * N + bn + c];
      f.z = B[(size_t)(k0 + kq + 2) * N + bn + c];
      f.w = B[(size_t)(k0 + kq + 3) * N + bn + c];
      *(v4bf*)&sBt[c][kq] = cvt4(f);
    }
    __syncthreads();

#pragma unroll
    for (int i = 0; i < 2; ++i) {
      const int m = wm + 16 * i + mr;
      v16bf af = frag_ld(&sA[m][8 * half], &sA[m][16 + 8 * half]);
#pragma unroll
      for (int j = 0; j < 2; ++j) {
        const int n = wn + 16 * j + mr;
        v16bf bfr = frag_ld(&sBt[n][8 * half], &sBt[n][16 + 8 * half]);
        acc[i][j] = WMMA_BF16(af, bfr, acc[i][j]);
      }
    }
    __syncthreads();
  }

#pragma unroll
  for (int i = 0; i < 2; ++i)
#pragma unroll
    for (int j = 0; j < 2; ++j) {
      int col  = bn + wn + 16 * j + mr;
      float bv = bias ? bias[col] : 0.f;
#pragma unroll
      for (int r = 0; r < 8; ++r) {
        int row = bm + wm + 16 * i + 8 * half + r;  // C layout: M = r + 8*half
        C[(size_t)row * N + col] = acc[i][j][r] + bv;
      }
    }
}

// ---------------------------------------------------------------------------
// Flash attention, dh=64, H=4, qkv row stride 768 ([q|k|v] per token).
// Block = 128 threads; 64 query rows per block (16 per wave); 32-key tiles.
// Online softmax: per-row (m,l) in LDS, O accumulators rescaled in registers.
// sK kept [key][feat] (feat innermost -> contiguous fragment chunks for QK^T);
// sV stored TRANSPOSED [feat][key] so PV B-fragments are contiguous too.
// ---------------------------------------------------------------------------
__global__ __launch_bounds__(128) void flash_attn_kernel(
    const float* __restrict__ qkv, float* __restrict__ out, int n) {
  __shared__ __align__(16) __bf16 sK[32][72];        // [key][feat], stride 144 B
  __shared__ __align__(16) __bf16 sVt[64][40];       // [feat][key], stride 80 B
  __shared__ __align__(16) __bf16 sPb[4][16][40];    // P tiles per wave
  __shared__ float sS[4][16][33];
  __shared__ float sM[4][16], sL[4][16], sAl[4][16];

  const int tid  = threadIdx.x;
  const int lane = tid & 31;
  const int wave = tid >> 5;
  const int head = blockIdx.y;
  const int q0   = blockIdx.x * 64 + wave * 16;
  const int half = lane >> 4;
  const int mr   = lane & 15;
  const float scale = 0.125f;  // 1/sqrt(64)

  // Q (16x64) as two 16x32 A-fragments, loaded once via float4 + packed cvt.
  v16bf qf[2];
#pragma unroll
  for (int g = 0; g < 2; ++g) {
    const float* qrow = qkv + (size_t)(q0 + mr) * 768 + head * 64 + 32 * g;
    float4 f0 = *(const float4*)(qrow + 8 * half);
    float4 f1 = *(const float4*)(qrow + 8 * half + 4);
    float4 f2 = *(const float4*)(qrow + 16 + 8 * half);
    float4 f3 = *(const float4*)(qrow + 16 + 8 * half + 4);
    v8bf lo = __builtin_shufflevector(cvt4(f0), cvt4(f1), 0, 1, 2, 3, 4, 5, 6, 7);
    v8bf hi = __builtin_shufflevector(cvt4(f2), cvt4(f3), 0, 1, 2, 3, 4, 5, 6, 7);
    qf[g] = __builtin_shufflevector(lo, hi, 0, 1, 2, 3, 4, 5, 6, 7,
                                    8, 9, 10, 11, 12, 13, 14, 15);
  }

  v8f o[4];
#pragma unroll
  for (int j = 0; j < 4; ++j)
#pragma unroll
    for (int r = 0; r < 8; ++r) o[j][r] = 0.f;

  if (lane < 16) { sM[wave][lane] = -1e30f; sL[wave][lane] = 0.f; }
  __syncthreads();

  for (int k0 = 0; k0 < n; k0 += 32) {
    if (k0 + 32 < n)  // prefetch next key/value rows
      __builtin_prefetch(qkv + (size_t)(k0 + 32 + (tid & 31)) * 768 + head * 64 + 256,
                         0, 1);
    // Stage K: [key][feat], float4 quads along feat (coalesced).
#pragma unroll
    for (int i = tid; i < 512; i += 128) {
      int r = i >> 4, qc = (i & 15) * 4;
      float4 f = *(const float4*)(qkv + (size_t)(k0 + r) * 768 + head * 64 + 256 + qc);
      *(v4bf*)&sK[r][qc] = cvt4(f);
    }
    // Stage V transposed: [feat][key]; per-t loads coalesced over feat c.
#pragma unroll
    for (int i = tid; i < 512; i += 128) {
      int c = i & 63, kq = (i >> 6) * 4;
      const float* vcol = qkv + (size_t)(k0 + kq) * 768 + head * 64 + 512 + c;
      float4 f;
      f.x = vcol[0];
      f.y = vcol[768];
      f.z = vcol[1536];
      f.w = vcol[2304];
      *(v4bf*)&sVt[c][kq] = cvt4(f);
    }
    __syncthreads();

    // S(16x32) = Q @ K^T : two 16-col tiles, dh=64 -> two k-steps each.
    v8f s[2];
#pragma unroll
    for (int j = 0; j < 2; ++j)
#pragma unroll
      for (int r = 0; r < 8; ++r) s[j][r] = 0.f;
#pragma unroll
    for (int j = 0; j < 2; ++j) {
      const int key = 16 * j + mr;
#pragma unroll
      for (int g = 0; g < 2; ++g) {
        v16bf kf = frag_ld(&sK[key][32 * g + 8 * half],
                           &sK[key][32 * g + 16 + 8 * half]);
        s[j] = WMMA_BF16(qf[g], kf, s[j]);
      }
    }
#pragma unroll
    for (int j = 0; j < 2; ++j)
#pragma unroll
      for (int r = 0; r < 8; ++r)
        sS[wave][8 * half + r][16 * j + mr] = s[j][r] * scale;
    __syncthreads();

    // Online-softmax row update; lanes 0-15 each own one row.
    if (lane < 16) {
      int r = lane;
      float mold = sM[wave][r];
      float mx   = mold;
      for (int c = 0; c < 32; ++c) mx = fmaxf(mx, sS[wave][r][c]);
      float al = expf(mold - mx);
      float ls = sL[wave][r] * al;
      for (int c = 0; c < 32; ++c) {
        float p = expf(sS[wave][r][c] - mx);
        sPb[wave][r][c] = (__bf16)p;
        ls += p;
      }
      sM[wave][r] = mx; sL[wave][r] = ls; sAl[wave][r] = al;
    }
    __syncthreads();

    // O = O*alpha + P @ V   (P 16x32 A-fragment; V fragments from sVt)
    v16bf pf = frag_ld(&sPb[wave][mr][8 * half], &sPb[wave][mr][16 + 8 * half]);
#pragma unroll
    for (int j = 0; j < 4; ++j) {
#pragma unroll
      for (int r = 0; r < 8; ++r) o[j][r] *= sAl[wave][8 * half + r];
      const int feat = 16 * j + mr;
      v16bf vf = frag_ld(&sVt[feat][8 * half], &sVt[feat][16 + 8 * half]);
      o[j] = WMMA_BF16(pf, vf, o[j]);
    }
    __syncthreads();
  }

#pragma unroll
  for (int j = 0; j < 4; ++j)
#pragma unroll
    for (int r = 0; r < 8; ++r) {
      int row = 8 * half + r;
      out[(size_t)(q0 + row) * 256 + head * 64 + 16 * j + mr] =
          o[j][r] / sL[wave][row];
    }
}

// ---------------------------------------------------------------------------
// Small VALU kernels (all memory-trivial at 23.3 TB/s)
// ---------------------------------------------------------------------------
__global__ __launch_bounds__(256) void embed_nodes_kernel(
    const float* __restrict__ x, const float* __restrict__ lap,
    const float* __restrict__ wn, const float* __restrict__ bn_,
    const float* __restrict__ wl, const float* __restrict__ bl,
    float* __restrict__ h, int N) {
  int i = blockIdx.x * 256 + threadIdx.x;
  if (i >= N * 256) return;
  int r = i >> 8, c = i & 255;
  float acc = bn_[c] + bl[c];
#pragma unroll
  for (int k = 0; k < 8; ++k) acc += x[r * 8 + k] * wn[k * 256 + c];
#pragma unroll
  for (int k = 0; k < 4; ++k) acc += lap[r * 4 + k] * wl[k * 256 + c];
  h[i] = acc;
}

__global__ __launch_bounds__(256) void embed_edges_kernel(
    const float* __restrict__ ea, const float* __restrict__ we,
    const float* __restrict__ be, float* __restrict__ e, int E) {
  int i = blockIdx.x * 256 + threadIdx.x;
  if (i >= E * 256) return;
  int r = i >> 8, c = i & 255;
  float acc = be[c];
#pragma unroll
  for (int k = 0; k < 4; ++k) acc += ea[r * 4 + k] * we[k * 256 + c];
  e[i] = acc;
}

__global__ __launch_bounds__(256) void layernorm_rows(
    const float* __restrict__ x, float* __restrict__ y) {
  __shared__ float red[256];
  int row = blockIdx.x, t = threadIdx.x;
  float v = x[(size_t)row * 256 + t];
  red[t] = v; __syncthreads();
  for (int s = 128; s > 0; s >>= 1) { if (t < s) red[t] += red[t + s]; __syncthreads(); }
  float mean = red[0] * (1.f / 256.f);
  __syncthreads();
  float c = v - mean;
  red[t] = c * c; __syncthreads();
  for (int s = 128; s > 0; s >>= 1) { if (t < s) red[t] += red[t + s]; __syncthreads(); }
  float var = red[0] * (1.f / 256.f);
  y[(size_t)row * 256 + t] = c * rsqrtf(var + 1e-5f);
}

__global__ __launch_bounds__(256) void bn_stats(
    const float* __restrict__ x, float* __restrict__ mean,
    float* __restrict__ var, int M, int C) {
  __shared__ float s1[256], s2[256];
  int col = blockIdx.x, t = threadIdx.x;
  float a = 0.f, b = 0.f;
  for (int r = t; r < M; r += 256) {
    float v = x[(size_t)r * C + col];
    a += v; b += v * v;
  }
  s1[t] = a; s2[t] = b; __syncthreads();
  for (int s = 128; s > 0; s >>= 1) {
    if (t < s) { s1[t] += s1[t + s]; s2[t] += s2[t + s]; }
    __syncthreads();
  }
  if (t == 0) {
    float m = s1[0] / (float)M;
    mean[col] = m;
    var[col]  = s2[0] / (float)M - m * m;  // biased variance (training mode)
  }
}

__global__ __launch_bounds__(256) void bn_apply_relu(
    float* __restrict__ x, const float* __restrict__ mean,
    const float* __restrict__ var, const float* __restrict__ g,
    const float* __restrict__ b, int total) {
  int i = blockIdx.x * 256 + threadIdx.x;
  if (i >= total) return;
  int c = i & 255;
  float v = (x[i] - mean[c]) * rsqrtf(var[c] + 1e-5f) * g[c] + b[c];
  x[i] = fmaxf(v, 0.f);
}

__global__ __launch_bounds__(256) void ew_add(
    const float* __restrict__ a, const float* __restrict__ b,
    float* __restrict__ o, int n) {
  int i = blockIdx.x * 256 + threadIdx.x;
  if (i < n) o[i] = a[i] + b[i];
}

// a = sigmoid(t1)*xn + t2
__global__ __launch_bounds__(256) void ew_swiglu_a(
    const float* __restrict__ t1, const float* __restrict__ xn,
    const float* __restrict__ t2, float* __restrict__ o, int n) {
  int i = blockIdx.x * 256 + threadIdx.x;
  if (i < n) o[i] = xn[i] / (1.f + expf(-t1[i])) + t2[i];
}

// b = silu(b12[:, :256]) * b12[:, 256:]
__global__ __launch_bounds__(256) void ew_silu_mul(
    const float* __restrict__ b12, float* __restrict__ o, int n) {
  int i = blockIdx.x * 256 + threadIdx.x;
  if (i >= n * 256) return;
  int r = i >> 8, c = i & 255;
  float u = b12[(size_t)r * 512 + c];
  float w = b12[(size_t)r * 512 + 256 + c];
  o[i] = (u / (1.f + expf(-u))) * w;
}

// out = x1 + sigmoid(g)*o2
__global__ __launch_bounds__(256) void ew_gate_add(
    const float* __restrict__ x1, const float* __restrict__ g,
    const float* __restrict__ o2, float* __restrict__ o, int n) {
  int i = blockIdx.x * 256 + threadIdx.x;
  if (i < n) o[i] = x1[i] + o2[i] / (1.f + expf(-g[i]));
}

__global__ __launch_bounds__(256) void zero_f(float* __restrict__ p, int n) {
  int i = blockIdx.x * 256 + threadIdx.x;
  if (i < n) p[i] = 0.f;
}

// m_in[e] = [h[dst[e]], h[src[e]], edge_feat[e]]   (PyG: src=idx[0], dst=idx[1])
__global__ __launch_bounds__(256) void gather_msg(
    const float* __restrict__ h, const float* __restrict__ e,
    const int* __restrict__ eidx, float* __restrict__ m, int E) {
  int i = blockIdx.x * 256 + threadIdx.x;
  if (i >= E * 768) return;
  int row = i / 768, c = i - row * 768;
  float v;
  if (c < 256)       v = h[(size_t)eidx[E + row] * 256 + c];
  else if (c < 512)  v = h[(size_t)eidx[row] * 256 + (c - 256)];
  else               v = e[(size_t)row * 256 + (c - 512)];
  m[i] = v;
}

__global__ __launch_bounds__(256) void scatter_add_edges(
    const float* __restrict__ m, const int* __restrict__ eidx,
    float* __restrict__ aggr, int E) {
  int i = blockIdx.x * 256 + threadIdx.x;
  if (i >= E * 256) return;
  int e = i >> 8, c = i & 255;
  atomicAdd(&aggr[(size_t)eidx[E + e] * 256 + c], m[i]);
}

__global__ __launch_bounds__(256) void concat_h_aggr(
    const float* __restrict__ h, const float* __restrict__ aggr,
    float* __restrict__ u, int N) {
  int i = blockIdx.x * 256 + threadIdx.x;
  if (i >= N * 512) return;
  int r = i >> 9, c = i & 511;
  u[i] = (c < 256) ? h[(size_t)r * 256 + c] : aggr[(size_t)r * 256 + (c - 256)];
}

__global__ __launch_bounds__(256) void pool_scatter(
    const float* __restrict__ h, const int* __restrict__ batch,
    float* __restrict__ sums, int N) {
  int i = blockIdx.x * 256 + threadIdx.x;
  if (i >= N * 256) return;
  int r = i >> 8, c = i & 255;
  atomicAdd(&sums[(size_t)batch[r] * 256 + c], h[i]);
}

__global__ __launch_bounds__(256) void pool_count(
    const int* __restrict__ batch, float* __restrict__ cnt, int N) {
  int i = blockIdx.x * 256 + threadIdx.x;
  if (i < N) atomicAdd(&cnt[batch[i]], 1.f);
}

__global__ __launch_bounds__(256) void pool_div(
    float* __restrict__ out, const float* __restrict__ cnt, int total) {
  int i = blockIdx.x * 256 + threadIdx.x;
  if (i < total) out[i] /= fmaxf(cnt[i >> 8], 1.f);
}

// ---------------------------------------------------------------------------
// Host-side orchestration
// ---------------------------------------------------------------------------
static inline void launch_gemm(const float* A, const float* B, const float* bias,
                               float* C, int M, int N, int K, hipStream_t s) {
  dim3 grid(M / 64, N / 64);
  gemm_bf16_wmma<<<grid, 128, 0, s>>>(A, B, bias, C, M, N, K);
}

static inline void launch_bn_relu(float* x, const float* g, const float* b,
                                  int M, float* mean, float* var, hipStream_t s) {
  bn_stats<<<256, 256, 0, s>>>(x, mean, var, M, 256);
  bn_apply_relu<<<(M * 256) / 256, 256, 0, s>>>(x, mean, var, g, b, M * 256);
}

struct Scratch {
  float *qkv, *attn, *x1, *xn, *t2, *t3, *b12, *mean, *var;
};

static void attn_block(float* xio, int n, const float* const* P /* 9 params */,
                       const Scratch& w, hipStream_t s) {
  const float *in_w = P[0], *in_b = P[1], *out_w = P[2], *out_b = P[3];
  const float *l1 = P[4], *l2 = P[5], *l3 = P[6], *s1 = P[7], *s2 = P[8];
  const int nel = n * 256;
  const int blks = nel / 256;

  launch_gemm(xio, in_w, in_b, w.qkv, n, 768, 256, s);            // qkv
  flash_attn_kernel<<<dim3(n / 64, 4), 128, 0, s>>>(w.qkv, w.attn, n);
  launch_gemm(w.attn, out_w, out_b, w.t2, n, 256, 256, s);        // out proj
  ew_add<<<blks, 256, 0, s>>>(xio, w.t2, w.x1, nel);              // x1 = x + mha
  layernorm_rows<<<n, 256, 0, s>>>(w.x1, w.xn);                   // xn = LN(x1)
  launch_gemm(w.xn, s1, nullptr, w.t2, n, 256, 256, s);           // xn@s1
  launch_gemm(w.xn, s2, nullptr, w.t3, n, 256, 256, s);           // xn@s2
  ew_swiglu_a<<<blks, 256, 0, s>>>(w.t2, w.xn, w.t3, w.t2, nel);  // a
  launch_gemm(w.t2, l1, nullptr, w.b12, n, 512, 256, s);          // a@l1 -> [n,512]
  ew_silu_mul<<<blks, 256, 0, s>>>(w.b12, w.t3, n);               // b
  launch_gemm(w.x1, l3, nullptr, w.xn, n, 256, 256, s);           // gate pre-act
  launch_gemm(w.t3, l2, nullptr, w.t2, n, 256, 256, s);           // b@l2
  ew_gate_add<<<blks, 256, 0, s>>>(w.x1, w.xn, w.t2, xio, nel);   // x + sig(g)*o
}

extern "C" void kernel_launch(void* const* d_in, const int* in_sizes, int n_in,
                              void* d_out, int out_size, void* d_ws, size_t ws_size,
                              hipStream_t stream) {
  (void)in_sizes; (void)n_in; (void)out_size; (void)ws_size;
  constexpr int N = 4096, E = 8192, G = 64, L = 4;

  const float* x_in   = (const float*)d_in[0];
  const float* lap    = (const float*)d_in[1];
  const float* eattr  = (const float*)d_in[2];
  const int*   eidx   = (const int*)d_in[3];
  const int*   batch  = (const int*)d_in[4];
  const float* w_node = (const float*)d_in[5];
  const float* b_node = (const float*)d_in[6];
  const float* w_edge = (const float*)d_in[7];
  const float* b_edge = (const float*)d_in[8];
  const float* w_lpe  = (const float*)d_in[9];
  const float* b_lpe  = (const float*)d_in[10];
  const float* na[9]; for (int k = 0; k < 9; ++k) na[k] = (const float*)d_in[11 + k];
  const float* ea[9]; for (int k = 0; k < 9; ++k) ea[k] = (const float*)d_in[20 + k];
  const float* msg_w1 = (const float*)d_in[29];
  const float* msg_b1 = (const float*)d_in[30];
  const float* msg_g1 = (const float*)d_in[31];
  const float* msg_be1= (const float*)d_in[32];
  const float* msg_w2 = (const float*)d_in[33];
  const float* msg_b2 = (const float*)d_in[34];
  const float* msg_g2 = (const float*)d_in[35];
  const float* msg_be2= (const float*)d_in[36];
  const float* upd_w1 = (const float*)d_in[37];
  const float* upd_b1 = (const float*)d_in[38];
  const float* upd_g1 = (const float*)d_in[39];
  const float* upd_be1= (const float*)d_in[40];
  const float* upd_w2 = (const float*)d_in[41];
  const float* upd_b2 = (const float*)d_in[42];
  const float* upd_g2 = (const float*)d_in[43];
  const float* upd_be2= (const float*)d_in[44];

  // Workspace bump allocation (~101 MB peak, sized for n=8192 attention).
  float* p = (float*)d_ws;
  auto alloc = [&](size_t nf) { float* r = p; p += nf; return r; };
  Scratch w;
  w.qkv  = alloc((size_t)E * 768);   // also reused as m_in [E,768]
  w.attn = alloc((size_t)E * 256);   // also reused as m1
  w.x1   = alloc((size_t)E * 256);   // also reused as m2
  w.xn   = alloc((size_t)E * 256);   // also reused as u1
  w.t2   = alloc((size_t)E * 256);
  w.t3   = alloc((size_t)E * 256);
  w.b12  = alloc((size_t)E * 512);   // also reused as u_in [N,512]
  float* h    = alloc((size_t)N * 256);
  float* e    = alloc((size_t)E * 256);
  float* aggr = alloc((size_t)N * 256);
  w.mean = alloc(256);
  w.var  = alloc(256);
  float* cnt = alloc(G);

  // ---- embeddings ----
  embed_nodes_kernel<<<(N * 256) / 256, 256, 0, stream>>>(
      x_in, lap, w_node, b_node, w_lpe, b_lpe, h, N);
  embed_edges_kernel<<<(E * 256) / 256, 256, 0, stream>>>(
      eattr, w_edge, b_edge, e, E);

  // ---- transformer blocks (flash attention + swiglu, all GEMMs on WMMA) ----
  attn_block(h, N, na, w, stream);
  attn_block(e, E, ea, w, stream);

  // ---- L rounds of message passing ----
  for (int l = 0; l < L; ++l) {
    gather_msg<<<(E * 768) / 256, 256, 0, stream>>>(h, e, eidx, w.qkv, E);
    launch_gemm(w.qkv, msg_w1 + (size_t)l * 768 * 256, msg_b1 + l * 256,
                w.attn, E, 256, 768, stream);
    launch_bn_relu(w.attn, msg_g1 + l * 256, msg_be1 + l * 256, E, w.mean, w.var, stream);
    launch_gemm(w.attn, msg_w2 + (size_t)l * 256 * 256, msg_b2 + l * 256,
                w.x1, E, 256, 256, stream);
    launch_bn_relu(w.x1, msg_g2 + l * 256, msg_be2 + l * 256, E, w.mean, w.var, stream);

    zero_f<<<(N * 256) / 256, 256, 0, stream>>>(aggr, N * 256);
    scatter_add_edges<<<(E * 256) / 256, 256, 0, stream>>>(w.x1, eidx, aggr, E);

    concat_h_aggr<<<(N * 512) / 256, 256, 0, stream>>>(h, aggr, w.b12, N);
    launch_gemm(w.b12, upd_w1 + (size_t)l * 512 * 256, upd_b1 + l * 256,
                w.xn, N, 256, 512, stream);
    launch_bn_relu(w.xn, upd_g1 + l * 256, upd_be1 + l * 256, N, w.mean, w.var, stream);
    launch_gemm(w.xn, upd_w2 + (size_t)l * 256 * 256, upd_b2 + l * 256,
                h, N, 256, 256, stream);
    launch_bn_relu(h, upd_g2 + l * 256, upd_be2 + l * 256, N, w.mean, w.var, stream);
  }

  // ---- global mean pool ----
  float* out = (float*)d_out;
  zero_f<<<(G * 256) / 256, 256, 0, stream>>>(out, G * 256);
  zero_f<<<1, 256, 0, stream>>>(cnt, G);
  pool_scatter<<<(N * 256) / 256, 256, 0, stream>>>(h, batch, out, N);
  pool_count<<<N / 256, 256, 0, stream>>>(batch, cnt, N);
  pool_div<<<(G * 256) / 256, 256, 0, stream>>>(out, cnt, G * 256);
}
